// AttentionLayer_62715112456647
// MI455X (gfx1250) — matrix operand order repaired and checked by reading, not compiled, not run on hardware
//
#include <hip/hip_runtime.h>
#include <hip/hip_bf16.h>

// ---------------------------------------------------------------------------
// AttentionLayer on MI455X (gfx1250): bf16 WMMA flash-attention pipeline.
//   B=16, V=1024, D=512, H=8, HD=64, MAXHOP=8
// Pipeline:
//   0) one-shot f32->bf16 conversion of x and all weights (kills in-loop cvts)
//   1) QKV projection, 16x64 register-blocked WMMA GEMM (V stored transposed)
//   2) flash attention w/ hop-bias, online softmax, LDS layout-transpose of P
//   3) output projection, 16x64 register-blocked, f32 result + bias
// ---------------------------------------------------------------------------

#define B_      16
#define V_      1024
#define D_      512
#define H_      8
#define HD_     64
#define MAXHOP_ 8

typedef __attribute__((ext_vector_type(16))) __bf16 v16bf;
typedef __attribute__((ext_vector_type(8)))  __bf16 v8bf;
typedef __attribute__((ext_vector_type(4)))  __bf16 v4bf;
typedef __attribute__((ext_vector_type(8)))  float  v8f;

// ---- WMMA wrapper ----------------------------------------------------------
__device__ __forceinline__ v8f wmma_bf16(v16bf a, v16bf b, v8f c) {
    return __builtin_amdgcn_wmma_f32_16x16x32_bf16(
        /*neg_a=*/false, a, /*neg_b=*/false, b,
        /*c_mod=*/(short)0, c, /*reuse_a=*/false, /*reuse_b=*/false);
}

// ---- A-matrix loader (16x32, MxK): lane holds row lane&15; -----------------
// K elements = { kbA..kbA+7 , 16+kbA..16+kbA+7 }, kbA = 8*(lane>=16)
__device__ __forceinline__ v16bf ldA_bf16(const __bf16* row, int kbA) {
    v8bf lo = *(const v8bf*)(row + kbA);
    v8bf hi = *(const v8bf*)(row + kbA + 16);
    v16bf r;
#pragma unroll
    for (int i = 0; i < 8; ++i) { r[i] = lo[i]; r[i + 8] = hi[i]; }
    return r;
}

// ---- B-matrix loader (32x16, KxN): lane holds col lane&15; -----------------
// K elements = { kbB/2..kbB/2+7 , 16+kbB/2..16+kbB/2+7 }, kbB = 16*(lane>=16)
__device__ __forceinline__ v16bf ldB_bf16(const __bf16* col, int kbB) {
    v8bf lo = *(const v8bf*)(col + (kbB >> 1));
    v8bf hi = *(const v8bf*)(col + (kbB >> 1) + 16);
    v16bf r;
#pragma unroll
    for (int i = 0; i < 8; ++i) { r[i] = lo[i]; r[i + 8] = hi[i]; }
    return r;
}

// ===========================================================================
// Kernel 0: vectorized f32 -> bf16 conversion (one shot; memory bound).
// ===========================================================================
__global__ __launch_bounds__(256) void cvt_bf16_kernel(
    const float* __restrict__ src, __bf16* __restrict__ dst, int n4)
{
    const int i = blockIdx.x * blockDim.x + threadIdx.x;
    if (i < n4) {
        const float4 v = *(const float4*)(src + (size_t)i * 4);
        v4bf o;
        o[0] = (__bf16)v.x; o[1] = (__bf16)v.y;
        o[2] = (__bf16)v.z; o[3] = (__bf16)v.w;
        *(v4bf*)(dst + (size_t)i * 4) = o;
    }
}

// ===========================================================================
// Kernel 1: fused QKV projection.  x(BV x D) @ W^T -> bf16, 16x64 per wave.
// Q,K stored row-major (BV x D); V stored transposed (b,h,hd,key).
// ===========================================================================
__global__ __launch_bounds__(256) void qkv_proj_kernel(
    const __bf16* __restrict__ xb,
    const __bf16* __restrict__ Wq, const __bf16* __restrict__ Wk,
    const __bf16* __restrict__ Wv,
    __bf16* __restrict__ qws, __bf16* __restrict__ kws,
    __bf16* __restrict__ vTws)
{
    const int wid   = blockIdx.x * 8 + (threadIdx.x >> 5);
    const int lane  = threadIdx.x & 31;
    const int ln    = lane & 15;
    const int hiw   = lane >> 4;
    const int kbA   = hiw << 3;    // 0 / 8
    const int kbB   = hiw << 4;    // 0 / 16
    const int half8 = hiw << 3;

    const int TILES = (B_ * V_ / 16) * (D_ / 64);   // 8192 per projection
    const int proj  = wid / TILES;
    const int r     = wid % TILES;
    const int m0    = (r / (D_ / 64)) * 16;         // row in BV
    const int n0    = (r % (D_ / 64)) * 64;         // col in D (one full head)

    const __bf16* W = (proj == 0) ? Wq : (proj == 1) ? Wk : Wv;
    const __bf16* xrow = xb + (size_t)(m0 + ln) * D_;
    const __bf16* wr0  = W + (size_t)(n0 + ln) * D_;   // B cols n0+t*16+ln

    v8f c0 = {}, c1 = {}, c2 = {}, c3 = {};
#pragma unroll 4
    for (int k0 = 0; k0 < D_; k0 += 32) {
        __builtin_prefetch(xrow + k0 + 64, 0, 1);
        const v16bf a = ldA_bf16(xrow + k0, kbA);
        c0 = wmma_bf16(a, ldB_bf16(wr0 + k0,                     kbB), c0);
        c1 = wmma_bf16(a, ldB_bf16(wr0 + (size_t)16 * D_ + k0,  kbB), c1);
        c2 = wmma_bf16(a, ldB_bf16(wr0 + (size_t)32 * D_ + k0,  kbB), c2);
        c3 = wmma_bf16(a, ldB_bf16(wr0 + (size_t)48 * D_ + k0,  kbB), c3);
    }
    v8f ct[4] = { c0, c1, c2, c3 };

    if (proj < 2) {
        __bf16* dst = (proj == 0) ? qws : kws;
#pragma unroll
        for (int t = 0; t < 4; ++t)
#pragma unroll
            for (int g = 0; g < 8; ++g)
                dst[(size_t)(m0 + half8 + g) * D_ + n0 + 16 * t + ln] =
                    (__bf16)ct[t][g];
    } else {
        // transposed store: vT[((b*H + h)*HD + hd)*V + key]; n0 is head-aligned
        const int b    = m0 / V_;
        const int key0 = (m0 % V_) + half8;     // 8 consecutive keys per lane
        const int h    = n0 >> 6;
#pragma unroll
        for (int t = 0; t < 4; ++t) {
            const int hd = 16 * t + ln;
            v8bf pk;
#pragma unroll
            for (int g = 0; g < 8; ++g) pk[g] = (__bf16)ct[t][g];
            *(v8bf*)(vTws + ((size_t)(b * H_ + h) * HD_ + hd) * V_ + key0) = pk;
        }
    }
}

// ===========================================================================
// Kernel 2: flash attention with hop-bias.  One wave per (b, h, 16-q tile).
// Online softmax over key blocks of 32; P layout-transposed through LDS.
// ===========================================================================
__global__ __launch_bounds__(256) void attn_kernel(
    const __bf16* __restrict__ qws, const __bf16* __restrict__ kws,
    const __bf16* __restrict__ vTws,
    const float* __restrict__ rpe, const int* __restrict__ hop,
    __bf16* __restrict__ aws)
{
    __shared__ float rpe_s[H_ * (MAXHOP_ + 1)];
    __shared__ __attribute__((aligned(16))) __bf16 Ps[8][16 * 32];

    for (int i = threadIdx.x; i < H_ * (MAXHOP_ + 1); i += blockDim.x)
        rpe_s[i] = rpe[i];
    __syncthreads();

    const int wslot = threadIdx.x >> 5;
    const int wid   = blockIdx.x * 8 + wslot;
    const int lane  = threadIdx.x & 31;
    const int ln    = lane & 15;
    const int hiw   = lane >> 4;
    const int kbA   = hiw << 3;
    const int kbB   = hiw << 4;
    const int half8 = hiw << 3;

    const int qtiles = V_ / 16;                 // 64
    const int b  = wid / (H_ * qtiles);
    const int rr = wid % (H_ * qtiles);
    const int h  = rr / qtiles;
    const int q0 = (rr % qtiles) * 16;

    const float* rpe_h = rpe_s + h * (MAXHOP_ + 1);
    const float scale  = 0.125f;                // 1/sqrt(HD)

    // Q tile: A-matrix regs, resident for whole loop
    const __bf16* qrow = qws + (size_t)(b * V_ + q0 + ln) * D_ + h * HD_;
    const v16bf aq0 = ldA_bf16(qrow,      kbA);       // hd  0..31
    const v16bf aq1 = ldA_bf16(qrow + 32, kbA);       // hd 32..63

    float M[8], S[8];
    v8f O0 = {}, O1 = {}, O2 = {}, O3 = {};
#pragma unroll
    for (int g = 0; g < 8; ++g) { M[g] = -1e30f; S[g] = 0.0f; }

    __bf16* myP = &Ps[wslot][0];

    for (int kb0 = 0; kb0 < V_; kb0 += 32) {
        // ---- scores: 16(q) x 32(keys) via 4 WMMAs ----
        const __bf16* krow0 = kws + (size_t)(b * V_ + kb0 + ln) * D_ + h * HD_;
        const __bf16* krow1 = krow0 + (size_t)16 * D_;
        v8f c0 = {}, c1 = {};
        c0 = wmma_bf16(aq0, ldB_bf16(krow0,      kbB), c0);
        c0 = wmma_bf16(aq1, ldB_bf16(krow0 + 32, kbB), c0);
        c1 = wmma_bf16(aq0, ldB_bf16(krow1,      kbB), c1);
        c1 = wmma_bf16(aq1, ldB_bf16(krow1 + 32, kbB), c1);

        // ---- bias + online softmax ----
        float p0[8], p1[8], alpha[8];
#pragma unroll
        for (int g = 0; g < 8; ++g) {
            const int qg = q0 + half8 + g;
            const float b0 = rpe_h[hop[(size_t)qg * V_ + kb0 + ln]];
            const float b1 = rpe_h[hop[(size_t)qg * V_ + kb0 + 16 + ln]];
            const float s0 = c0[g] * scale + b0;
            const float s1 = c1[g] * scale + b1;
            float t = fmaxf(s0, s1);
            t = fmaxf(t, __shfl_xor(t, 1, 16));
            t = fmaxf(t, __shfl_xor(t, 2, 16));
            t = fmaxf(t, __shfl_xor(t, 4, 16));
            t = fmaxf(t, __shfl_xor(t, 8, 16));
            const float nM = fmaxf(M[g], t);
            alpha[g] = __expf(M[g] - nM);
            M[g] = nM;
            p0[g] = __expf(s0 - nM);
            p1[g] = __expf(s1 - nM);
            float rs = p0[g] + p1[g];
            rs += __shfl_xor(rs, 1, 16);
            rs += __shfl_xor(rs, 2, 16);
            rs += __shfl_xor(rs, 4, 16);
            rs += __shfl_xor(rs, 8, 16);
            S[g] = S[g] * alpha[g] + rs;
        }
#pragma unroll
        for (int g = 0; g < 8; ++g) {
            O0[g] *= alpha[g]; O1[g] *= alpha[g];
            O2[g] *= alpha[g]; O3[g] *= alpha[g];
        }

        // ---- C-layout -> A-layout transpose of P through LDS ----
#pragma unroll
        for (int g = 0; g < 8; ++g) {
            const int m = half8 + g;
            myP[m * 32 + ln]      = (__bf16)p0[g];
            myP[m * 32 + 16 + ln] = (__bf16)p1[g];
        }
        __syncthreads();
        const v16bf aP = ldA_bf16(myP + (size_t)ln * 32, kbA);

        // ---- O(16x64) += P(16x32) @ V(32x64): 4 WMMAs ----
        const __bf16* vbase =
            vTws + ((size_t)(b * H_ + h) * HD_ + ln) * V_ + kb0;
        O0 = wmma_bf16(aP, ldB_bf16(vbase,                   kbB), O0);
        O1 = wmma_bf16(aP, ldB_bf16(vbase + (size_t)16 * V_, kbB), O1);
        O2 = wmma_bf16(aP, ldB_bf16(vbase + (size_t)32 * V_, kbB), O2);
        O3 = wmma_bf16(aP, ldB_bf16(vbase + (size_t)48 * V_, kbB), O3);
        __syncthreads();
    }

    // ---- normalize + store bf16 attention output (row-major BV x D) ----
    __bf16* orow = aws + (size_t)(b * V_ + q0) * D_ + h * HD_;
#pragma unroll
    for (int g = 0; g < 8; ++g) {
        const float inv = 1.0f / S[g];
        __bf16* p = orow + (size_t)(half8 + g) * D_;
        p[ln]      = (__bf16)(O0[g] * inv);
        p[16 + ln] = (__bf16)(O1[g] * inv);
        p[32 + ln] = (__bf16)(O2[g] * inv);
        p[48 + ln] = (__bf16)(O3[g] * inv);
    }
}

// ===========================================================================
// Kernel 3: output projection  attn(BV x D) @ Wo^T + bo -> f32, 16x64/wave
// ===========================================================================
__global__ __launch_bounds__(256) void out_proj_kernel(
    const __bf16* __restrict__ aws,
    const __bf16* __restrict__ Wo, const float* __restrict__ bo,
    float* __restrict__ out)
{
    const int wid   = blockIdx.x * 8 + (threadIdx.x >> 5);
    const int lane  = threadIdx.x & 31;
    const int ln    = lane & 15;
    const int hiw   = lane >> 4;
    const int kbA   = hiw << 3;
    const int kbB   = hiw << 4;
    const int half8 = hiw << 3;

    const int m0 = (wid / (D_ / 64)) * 16;
    const int n0 = (wid % (D_ / 64)) * 64;

    const __bf16* arow = aws + (size_t)(m0 + ln) * D_;
    const __bf16* wr0  = Wo + (size_t)(n0 + ln) * D_;

    v8f c0 = {}, c1 = {}, c2 = {}, c3 = {};
#pragma unroll 4
    for (int k0 = 0; k0 < D_; k0 += 32) {
        const v16bf a = ldA_bf16(arow + k0, kbA);
        c0 = wmma_bf16(a, ldB_bf16(wr0 + k0,                    kbB), c0);
        c1 = wmma_bf16(a, ldB_bf16(wr0 + (size_t)16 * D_ + k0, kbB), c1);
        c2 = wmma_bf16(a, ldB_bf16(wr0 + (size_t)32 * D_ + k0, kbB), c2);
        c3 = wmma_bf16(a, ldB_bf16(wr0 + (size_t)48 * D_ + k0, kbB), c3);
    }
    v8f ct[4] = { c0, c1, c2, c3 };
#pragma unroll
    for (int t = 0; t < 4; ++t) {
        const float bias = bo[n0 + 16 * t + ln];
#pragma unroll
        for (int g = 0; g < 8; ++g)
            out[(size_t)(m0 + half8 + g) * D_ + n0 + 16 * t + ln] =
                ct[t][g] + bias;
    }
}

// ===========================================================================
// Launcher
// ===========================================================================
extern "C" void kernel_launch(void* const* d_in, const int* in_sizes, int n_in,
                              void* d_out, int out_size, void* d_ws, size_t ws_size,
                              hipStream_t stream)
{
    (void)in_sizes; (void)n_in; (void)out_size; (void)ws_size;

    const float* x   = (const float*)d_in[0];
    const float* Wq  = (const float*)d_in[1];
    const float* Wk  = (const float*)d_in[2];
    const float* Wv  = (const float*)d_in[3];
    const float* Wo  = (const float*)d_in[4];
    const float* bo  = (const float*)d_in[5];
    const float* rpe = (const float*)d_in[6];
    const int*   hop = (const int*)d_in[7];
    float* out = (float*)d_out;

    const size_t XB = (size_t)B_ * V_ * D_ * sizeof(__bf16);   // 16 MiB
    const size_t WB = (size_t)D_ * D_ * sizeof(__bf16);        // 512 KiB
    char* ws = (char*)d_ws;
    __bf16* xb  = (__bf16*)(ws);
    __bf16* wqb = (__bf16*)(ws + XB);
    __bf16* wkb = (__bf16*)(ws + XB + WB);
    __bf16* wvb = (__bf16*)(ws + XB + 2 * WB);
    __bf16* wob = (__bf16*)(ws + XB + 3 * WB);
    __bf16* qws = (__bf16*)(ws + XB + 4 * WB);
    __bf16* kws = (__bf16*)(ws + 2 * XB + 4 * WB);
    __bf16* vT  = (__bf16*)(ws + 3 * XB + 4 * WB);
    __bf16* aws = (__bf16*)(ws + 4 * XB + 4 * WB);

    const dim3 blk(256);

    // 0) one-shot f32 -> bf16 conversions
    const int nx4 = B_ * V_ * D_ / 4;          // 2,097,152
    const int nw4 = D_ * D_ / 4;               // 65,536
    cvt_bf16_kernel<<<dim3((nx4 + 255) / 256), blk, 0, stream>>>(x, xb, nx4);
    cvt_bf16_kernel<<<dim3((nw4 + 255) / 256), blk, 0, stream>>>(Wq, wqb, nw4);
    cvt_bf16_kernel<<<dim3((nw4 + 255) / 256), blk, 0, stream>>>(Wk, wkb, nw4);
    cvt_bf16_kernel<<<dim3((nw4 + 255) / 256), blk, 0, stream>>>(Wv, wvb, nw4);
    cvt_bf16_kernel<<<dim3((nw4 + 255) / 256), blk, 0, stream>>>(Wo, wob, nw4);

    // 1) QKV projection: 3 * 1024 * 8 wave-tasks, 8 waves/block
    qkv_proj_kernel<<<dim3(3 * (B_ * V_ / 16) * (D_ / 64) / 8), blk, 0, stream>>>(
        xb, wqb, wkb, wvb, qws, kws, vT);

    // 2) attention: B*H*(V/16) wave tasks, 8 waves/block
    attn_kernel<<<dim3(B_ * H_ * (V_ / 16) / 8), blk, 0, stream>>>(
        qws, kws, vT, rpe, hop, aws);

    // 3) output projection: 1024*8 wave tasks, 8 waves/block
    out_proj_kernel<<<dim3((B_ * V_ / 16) * (D_ / 64) / 8), blk, 0, stream>>>(
        aws, wob, bo, out);
}
